// StateRNN_73632919322882
// MI455X (gfx1250) — matrix operand order repaired and checked by reading, not compiled, not run on hardware
//
#include <hip/hip_runtime.h>
#include <hip/hip_bf16.h>

// ---------------------------------------------------------------------------
// Types for WMMA fragments (CDNA5 / gfx1250, wave32)
// ---------------------------------------------------------------------------
typedef __attribute__((ext_vector_type(16))) __bf16 bf16x16;
typedef __attribute__((ext_vector_type(8)))  __bf16 bf16x8;
typedef __attribute__((ext_vector_type(8)))  float  f32x8;

// Problem constants
#define BB   64
#define TT   2048
#define NIN  48
#define HH   256
#define KPAD 320          // 48 x-cols + 16 zero pad + 256 h-cols
#define LAT  32

// A-fragment: 16x32 bf16 tile from a row-major source.
// Lane half h holds K = 16h..16h+15 (contiguous, matching frag_b).
__device__ __forceinline__ bf16x16 frag_a(const __bf16* row_ptr, int kbase) {
    bf16x8 lo = *(const bf16x8*)(row_ptr + kbase);
    bf16x8 hi = *(const bf16x8*)(row_ptr + kbase + 8);
    bf16x16 r;
#pragma unroll
    for (int i = 0; i < 8; ++i) { r[i] = lo[i]; r[i + 8] = hi[i]; }
    return r;
}

// B-fragment: 32x16 bf16 from N-major (transposed) weights; lane half h holds
// K = 16h..16h+15, i.e. 16 contiguous bf16 (32 bytes) of column n.
__device__ __forceinline__ bf16x16 frag_b(const __bf16* p) {
    return *(const bf16x16*)p;
}

// Branch-free transcendentals: v_exp_f32 + v_rcp_f32, no exec-mask divergence.
// Saturation: x->+inf: exp->inf, rcp->0 => 1 ; x->-inf: exp->0, rcp(1)=1 => -1.
__device__ __forceinline__ float sigmoidf_(float x) {
    return __builtin_amdgcn_rcpf(1.0f + __expf(-x));
}
__device__ __forceinline__ float tanhf_(float x) {
    return 1.0f - 2.0f * __builtin_amdgcn_rcpf(__expf(2.0f * x) + 1.0f);
}

// ---------------------------------------------------------------------------
// Weight preparation: f32 -> bf16, transposed to N-major so B-fragments are
// contiguous 32B loads.
// ---------------------------------------------------------------------------
__global__ void build_WT_kernel(const float* __restrict__ wx,
                                const float* __restrict__ wh,
                                __bf16* __restrict__ WT) {
    int idx = blockIdx.x * blockDim.x + threadIdx.x;   // n*KPAD + k
    if (idx >= 1024 * KPAD) return;
    int n = idx / KPAD, k = idx % KPAD;
    float v = 0.0f;
    if (k < NIN)      v = wx[k * 1024 + n];
    else if (k >= 64) v = wh[(k - 64) * 1024 + n];
    WT[idx] = (__bf16)v;
}

__global__ void transpose_bf16_kernel(const float* __restrict__ src,
                                      __bf16* __restrict__ dst, int K, int N) {
    int idx = blockIdx.x * blockDim.x + threadIdx.x;   // n*K + k
    if (idx >= K * N) return;
    int n = idx / K, k = idx % K;
    dst[idx] = (__bf16)src[k * N + n];
}

// ---------------------------------------------------------------------------
// LSTM scan: single workgroup, 512 threads = 16 waves, state resident in LDS.
// Per step: z = [x_t | 0 | h] @ WT^T  (K=320, all four gates), then gates.
// Wave w owns gate-column tile nt=w (16 cols) and ALL 4 batch tiles, so each
// B-fragment is shared by 4 WMMAs and WT is streamed from L2 exactly once
// per step. Three s_barriers per step; masks for t+1 pre-applied at step t.
// ---------------------------------------------------------------------------
__global__ __launch_bounds__(512) void lstm_scan_kernel(
        const float* __restrict__ seq,      // [B,T,NIN]
        const float* __restrict__ mask,     // [B,T]
        const float* __restrict__ states,   // [B,2H]
        const float* __restrict__ bias,     // [4H]
        const __bf16* __restrict__ WT,      // [1024][KPAD] bf16, N-major
        __bf16* __restrict__ hs,            // [T,B,H] bf16 (workspace)
        float* __restrict__ states_out)     // [B,2H] (tail of d_out)
{
    extern __shared__ char smem[];
    float*  c_s  = (float*)smem;                                 // 64 KB
    __bf16* xaug = (__bf16*)(smem + BB * HH * 4);                // 40 KB
    float*  b_s  = (float*)(smem + BB * HH * 4 + BB * KPAD * 2); //  4 KB
    float*  mn_s = b_s + 4 * HH;                                 // 2*64 f32

    const int tid  = threadIdx.x;
    const int lane = tid & 31;
    const int wave = tid >> 5;          // 0..15  -> nt
    const int half = lane >> 4;         // 0/1
    const int lr   = lane & 15;

    // ---- init: bias, masked c0/h0, zero K-padding --------------------------
    for (int i = tid; i < 4 * HH; i += 512) b_s[i] = bias[i];
    for (int i = tid; i < BB * HH; i += 512) {
        int b = i >> 8, col = i & 255;
        float m = 1.0f - mask[b * TT + 0];
        c_s[i] = states[b * 2 * HH + col] * m;
        xaug[b * KPAD + 64 + col] = (__bf16)(states[b * 2 * HH + HH + col] * m);
    }
    for (int i = tid; i < BB * 16; i += 512) {
        int b = i >> 4;
        xaug[b * KPAD + NIN + (i & 15)] = (__bf16)0.0f;
    }
    __syncthreads();

    const int nt  = wave;
    const int col = nt * 16 + lr;

    for (int t = 0; t < TT; ++t) {
        // ---- phase 1: stage x_t, prefetch x_{t+1}, next-step mask ----------
        for (int e = tid; e < BB * NIN; e += 512) {
            int b = e / NIN, c = e % NIN;
            xaug[b * KPAD + c] = (__bf16)seq[(b * TT + t) * NIN + c];
        }
        if (tid < BB) {
            mn_s[(t & 1) * BB + tid] =
                (t + 1 < TT) ? (1.0f - mask[tid * TT + t + 1]) : 1.0f;
            if (t + 1 < TT)   // warm L2/L0 for next step's x rows
                __builtin_prefetch(seq + (tid * TT + t + 1) * NIN, 0, 3);
        }
        __syncthreads();

        // ---- phase 2: WMMA — 4 batch tiles x 4 gates per wave --------------
        f32x8 acc[4][4];
#pragma unroll
        for (int mt = 0; mt < 4; ++mt)
#pragma unroll
            for (int g = 0; g < 4; ++g)
#pragma unroll
                for (int j = 0; j < 8; ++j) acc[mt][g][j] = 0.0f;

#pragma unroll 1
        for (int kt = 0; kt < KPAD / 32; ++kt) {
            const int kbase = kt * 32 + half * 16;
            bf16x16 a[4];
#pragma unroll
            for (int mt = 0; mt < 4; ++mt)
                a[mt] = frag_a(xaug + (mt * 16 + lr) * KPAD, kbase);
#pragma unroll
            for (int g = 0; g < 4; ++g) {
                bf16x16 bfrag = frag_b(
                    WT + (g * HH + nt * 16 + lr) * KPAD + kt * 32 + half * 16);
#pragma unroll
                for (int mt = 0; mt < 4; ++mt)
                    acc[mt][g] = __builtin_amdgcn_wmma_f32_16x16x32_bf16(
                        false, a[mt], false, bfrag, (short)0, acc[mt][g],
                        false, false);
            }
        }
        __syncthreads();

        // ---- phase 3: gates / state update / h store -----------------------
        const bool  last = (t == TT - 1);
        const float bi = b_s[col];
        const float bf = b_s[HH + col];
        const float bo = b_s[2 * HH + col];
        const float bu = b_s[3 * HH + col];
#pragma unroll
        for (int mt = 0; mt < 4; ++mt) {
#pragma unroll
            for (int j = 0; j < 8; ++j) {
                const int m_row = mt * 16 + half * 8 + j;
                float ig = sigmoidf_(acc[mt][0][j] + bi);
                float fg = sigmoidf_(acc[mt][1][j] + bf);
                float og = sigmoidf_(acc[mt][2][j] + bo);
                float ug = tanhf_(acc[mt][3][j] + bu);
                float cn = fg * c_s[m_row * HH + col] + ig * ug;
                float hn = og * tanhf_(cn);
                hs[(t * BB + m_row) * HH + col] = (__bf16)hn;   // unmasked h_t
                if (last) {
                    states_out[m_row * 2 * HH + col]      = cn;
                    states_out[m_row * 2 * HH + HH + col] = hn;
                } else {
                    float mn = mn_s[(t & 1) * BB + m_row];
                    c_s[m_row * HH + col]         = cn * mn;
                    xaug[m_row * KPAD + 64 + col] = (__bf16)(hn * mn);
                }
            }
        }
        __syncthreads();
    }
}

// ---------------------------------------------------------------------------
// Fused MLP head: per WG, 128 rows through relu(x@w1+b1)@... with d1/d2 in
// LDS. Each of the 8 waves owns one 16-row M-tile; A-fragments are preloaded
// into registers once per layer and reused across every N-tile.
// Flat output index (row*32 + col) reproduces the reference's reshape exactly.
// ---------------------------------------------------------------------------
__global__ __launch_bounds__(256) void mlp_head_kernel(
        const __bf16* __restrict__ hs,    // [T*B, 256] bf16
        const __bf16* __restrict__ w1T,   // [256][256]
        const float*  __restrict__ b1,
        const __bf16* __restrict__ w2T,   // [128][256]
        const float*  __restrict__ b2,
        const __bf16* __restrict__ w3T,   // [32][128]
        const float*  __restrict__ b3,
        float* __restrict__ out)          // [T*B, 32] flat
{
    extern __shared__ char smem[];
    __bf16* d1s = (__bf16*)smem;                       // 128*256*2 = 64 KB
    __bf16* d2s = (__bf16*)(smem + 128 * 256 * 2);     // 128*128*2 = 32 KB

    const int tid  = threadIdx.x;
    const int lane = tid & 31;
    const int wave = tid >> 5;        // 0..7 -> M-tile
    const int half = lane >> 4;
    const int lr   = lane & 15;
    const int rowbase = blockIdx.x * 128;
    const int mrow = wave * 16 + lr;  // fragment row within the 128-row block

    // ---- layer 1: [128,256] @ [256,256], relu ------------------------------
    {
        bf16x16 a[8];
        const __bf16* arow = hs + (size_t)(rowbase + mrow) * HH;
#pragma unroll
        for (int kt = 0; kt < 8; ++kt) a[kt] = frag_a(arow, kt * 32 + half * 16);
#pragma unroll 1
        for (int nt = 0; nt < 16; ++nt) {
            f32x8 acc;
#pragma unroll
            for (int j = 0; j < 8; ++j) acc[j] = 0.0f;
#pragma unroll
            for (int kt = 0; kt < 8; ++kt) {
                bf16x16 b = frag_b(w1T + (nt * 16 + lr) * 256 + kt * 32 + half * 16);
                acc = __builtin_amdgcn_wmma_f32_16x16x32_bf16(
                    false, a[kt], false, b, (short)0, acc, false, false);
            }
            const int col = nt * 16 + lr;
            const float bb = b1[col];
#pragma unroll
            for (int j = 0; j < 8; ++j) {
                float v = acc[j] + bb;
                d1s[(wave * 16 + half * 8 + j) * 256 + col] =
                    (__bf16)(v > 0.0f ? v : 0.0f);
            }
        }
    }
    __syncthreads();

    // ---- layer 2: [128,256] @ [256,128], relu ------------------------------
    {
        bf16x16 a[8];
        const __bf16* arow = d1s + mrow * 256;
#pragma unroll
        for (int kt = 0; kt < 8; ++kt) a[kt] = frag_a(arow, kt * 32 + half * 16);
#pragma unroll 1
        for (int nt = 0; nt < 8; ++nt) {
            f32x8 acc;
#pragma unroll
            for (int j = 0; j < 8; ++j) acc[j] = 0.0f;
#pragma unroll
            for (int kt = 0; kt < 8; ++kt) {
                bf16x16 b = frag_b(w2T + (nt * 16 + lr) * 256 + kt * 32 + half * 16);
                acc = __builtin_amdgcn_wmma_f32_16x16x32_bf16(
                    false, a[kt], false, b, (short)0, acc, false, false);
            }
            const int col = nt * 16 + lr;
            const float bb = b2[col];
#pragma unroll
            for (int j = 0; j < 8; ++j) {
                float v = acc[j] + bb;
                d2s[(wave * 16 + half * 8 + j) * 128 + col] =
                    (__bf16)(v > 0.0f ? v : 0.0f);
            }
        }
    }
    __syncthreads();

    // ---- layer 3: [128,128] @ [128,32] -> global ---------------------------
    {
        bf16x16 a[4];
        const __bf16* arow = d2s + mrow * 128;
#pragma unroll
        for (int kt = 0; kt < 4; ++kt) a[kt] = frag_a(arow, kt * 32 + half * 16);
#pragma unroll 1
        for (int nt = 0; nt < 2; ++nt) {
            f32x8 acc;
#pragma unroll
            for (int j = 0; j < 8; ++j) acc[j] = 0.0f;
#pragma unroll
            for (int kt = 0; kt < 4; ++kt) {
                bf16x16 b = frag_b(w3T + (nt * 16 + lr) * 128 + kt * 32 + half * 16);
                acc = __builtin_amdgcn_wmma_f32_16x16x32_bf16(
                    false, a[kt], false, b, (short)0, acc, false, false);
            }
            const int col = nt * 16 + lr;
            const float bb = b3[col];
#pragma unroll
            for (int j = 0; j < 8; ++j) {
                const int R = rowbase + wave * 16 + half * 8 + j; // time-major row
                out[(size_t)R * LAT + col] = acc[j] + bb;
            }
        }
    }
}

// ---------------------------------------------------------------------------
// Host-side launcher
// ---------------------------------------------------------------------------
extern "C" void kernel_launch(void* const* d_in, const int* in_sizes, int n_in,
                              void* d_out, int out_size, void* d_ws, size_t ws_size,
                              hipStream_t stream) {
    const float* seq    = (const float*)d_in[0];   // [64,2048,48]
    const float* rmask  = (const float*)d_in[1];   // [64,2048]
    const float* states = (const float*)d_in[2];   // [64,512]
    const float* wx     = (const float*)d_in[3];   // [48,1024]
    const float* wh     = (const float*)d_in[4];   // [256,1024]
    const float* bias   = (const float*)d_in[5];   // [1024]
    const float* w1     = (const float*)d_in[6];   // [256,256]
    const float* b1     = (const float*)d_in[7];   // [256]
    const float* w2     = (const float*)d_in[8];   // [256,128]
    const float* b2     = (const float*)d_in[9];   // [128]
    const float* w3     = (const float*)d_in[10];  // [128,32]
    const float* b3     = (const float*)d_in[11];  // [32]

    float* out = (float*)d_out;                    // [64*2048*32] + [64*512]
    float* states_out = out + (size_t)BB * TT * LAT;

    // Workspace layout (bytes)
    char* ws = (char*)d_ws;
    __bf16* WT  = (__bf16*)(ws + 0);                      // 1024*320*2 = 655360
    __bf16* w1T = (__bf16*)(ws + 655360);                 // 256*256*2  = 131072
    __bf16* w2T = (__bf16*)(ws + 786432);                 // 128*256*2  =  65536
    __bf16* w3T = (__bf16*)(ws + 851968);                 // 32*128*2   =   8192
    __bf16* hsb = (__bf16*)(ws + 860160);                 // 2048*64*256*2 = 64 MiB

    // 1) weight conversion / transposition to bf16 N-major
    build_WT_kernel<<<(1024 * KPAD + 255) / 256, 256, 0, stream>>>(wx, wh, WT);
    transpose_bf16_kernel<<<(256 * 256 + 255) / 256, 256, 0, stream>>>(w1, w1T, 256, 256);
    transpose_bf16_kernel<<<(256 * 128 + 255) / 256, 256, 0, stream>>>(w2, w2T, 256, 128);
    transpose_bf16_kernel<<<(128 * 32 + 255) / 256, 256, 0, stream>>>(w3, w3T, 128, 32);

    // 2) sequential LSTM scan — one workgroup, state in LDS
    size_t scan_lds = (size_t)BB * HH * 4 + (size_t)BB * KPAD * 2
                    + 4 * HH * 4 + 2 * BB * 4;
    lstm_scan_kernel<<<1, 512, scan_lds, stream>>>(
        seq, rmask, states, bias, WT, hsb, states_out);

    // 3) fused MLP head over all 131072 rows
    size_t mlp_lds = (size_t)128 * 256 * 2 + (size_t)128 * 128 * 2;
    mlp_head_kernel<<<(BB * TT) / 128, 256, mlp_lds, stream>>>(
        hsb, w1T, b1, w2T, b2, w3T, b3, out);
}
